// VectorQuantizer_17179869577
// MI455X (gfx1250) — compile-verified, matrix operand order checked by reading, not run on hardware
//
#include <hip/hip_runtime.h>

// ---------------------------------------------------------------------------
// VectorQuantizer for MI455X (gfx1250, wave32, WMMA + async-to-LDS).
//   dist[n,k] = ||c_k||^2 - 2 * <z_n, c_k>
// GEMM via v_wmma_f32_16x16x32_bf16, fp32 = hi(bf16)+lo(bf16) 3-term split.
// Codebook pre-converted once to padded bf16 hi/lo planes in d_ws; staged to
// LDS with GLOBAL_LOAD_ASYNC_TO_LDS_B128 (double-buffered, s_wait_asynccnt
// pipelined). Argmin update is software-pipelined one tile behind the WMMAs
// so the WMMA->VALU hazard and the argmin VALU co-execute with the next
// tile's (TRANS-class) WMMAs.
// Outputs (flat, float): z_q_st [N*D] | vq_loss [1] | indices [N] | z_q [N*D]
// ---------------------------------------------------------------------------

typedef __attribute__((ext_vector_type(16))) __bf16 v16bf;
typedef __attribute__((ext_vector_type(8)))  float  v8f;

#define N_TOT      262144
#define K_CB       1024
#define D_DIM      64
#define ROWSTRIDE  72            // padded row stride (bf16 elems): 144B, 16B-aligned, bank-conflict-free
#define CHUNK_K    64            // codebook rows per LDS chunk (double buffered)
#define NCHUNK     (K_CB / CHUNK_K)       // 16
#define BLK_THREADS 512
#define WAVES_PER_BLK 16
#define ROWS_PER_BLK 256                  // 16 rows per wave
#define NBLK       (N_TOT / ROWS_PER_BLK) // 1024

__device__ __forceinline__ unsigned short f2bf(float x) {   // RNE fp32->bf16
    unsigned u = __float_as_uint(x);
    u += 0x7FFFu + ((u >> 16) & 1u);
    return (unsigned short)(u >> 16);
}
__device__ __forceinline__ float bf2f(unsigned short s) {
    return __uint_as_float(((unsigned)s) << 16);
}

union Frag {                      // one WMMA 16-bit operand (8 VGPRs / lane)
    v16bf bf;
    unsigned short s[16];
    uint4 q[2];
};

// A fragment (z rows), ISA 16-bit A 16x32 layout: lane-half offset dh in {0,8};
// elems 0..7 -> d = dbase+dh+j, elems 8..15 -> d = dbase+16+dh+j.
__device__ __forceinline__ void build_a_frag(const float* __restrict__ zrow,
                                             int dbase, int dh,
                                             Frag& hi, Frag& lo) {
#pragma unroll
    for (int h = 0; h < 2; ++h) {
        const float* p = zrow + dbase + h * 16 + dh;
#pragma unroll
        for (int j = 0; j < 8; j += 4) {
            float4 v = *(const float4*)(p + j);
            float xs[4] = {v.x, v.y, v.z, v.w};
#pragma unroll
            for (int e = 0; e < 4; ++e) {
                unsigned short hb = f2bf(xs[e]);
                hi.s[h * 8 + j + e] = hb;
                lo.s[h * 8 + j + e] = f2bf(xs[e] - bf2f(hb));
            }
        }
    }
}

__device__ __forceinline__ Frag load_b_frag(const unsigned short* plane,
                                            int kc, int dbase, int dh) {
    Frag f;
    f.q[0] = *(const uint4*)&plane[kc * ROWSTRIDE + dbase + dh];       // ds_load_b128
    f.q[1] = *(const uint4*)&plane[kc * ROWSTRIDE + dbase + 16 + dh];  // ds_load_b128
    return f;
}

// Kernel 0: one-time codebook prep: ||c_k||^2 and padded bf16 hi/lo planes.
__global__ void vq_prep_kernel(const float* __restrict__ cb,
                               float* __restrict__ csq,
                               unsigned short* __restrict__ g_hi,
                               unsigned short* __restrict__ g_lo) {
    int k = blockIdx.x * blockDim.x + threadIdx.x;
    if (k < K_CB) {
        const float* row = cb + (size_t)k * D_DIM;
        float s = 0.f;
#pragma unroll
        for (int j = 0; j < D_DIM; j += 4) {
            float4 v = *(const float4*)(row + j);
            float xs[4] = {v.x, v.y, v.z, v.w};
            unsigned hs[4], ls[4];
#pragma unroll
            for (int e = 0; e < 4; ++e) {
                s += xs[e] * xs[e];
                unsigned short hb = f2bf(xs[e]);
                hs[e] = hb;
                ls[e] = f2bf(xs[e] - bf2f(hb));
            }
            uint2 ph; ph.x = hs[0] | (hs[1] << 16); ph.y = hs[2] | (hs[3] << 16);
            uint2 pl; pl.x = ls[0] | (ls[1] << 16); pl.y = ls[2] | (ls[3] << 16);
            *(uint2*)&g_hi[(size_t)k * ROWSTRIDE + j] = ph;
            *(uint2*)&g_lo[(size_t)k * ROWSTRIDE + j] = pl;
        }
        csq[k] = s;
    }
}

// Kernel 1: WMMA distances + argmin + gather + outputs + per-block loss partial
__global__ __launch_bounds__(BLK_THREADS) void vq_main_kernel(
        const float* __restrict__ z, const float* __restrict__ cb,
        const float* __restrict__ csq,
        const unsigned short* __restrict__ g_hi,
        const unsigned short* __restrict__ g_lo,
        float* __restrict__ out, float* __restrict__ partials) {
    __shared__ __align__(16) unsigned short sh_hi[2][CHUNK_K * ROWSTRIDE];
    __shared__ __align__(16) unsigned short sh_lo[2][CHUNK_K * ROWSTRIDE];
    __shared__ float sh_csq[2][CHUNK_K];
    __shared__ int   sh_idx[WAVES_PER_BLK * 16];
    __shared__ float sh_wsum[WAVES_PER_BLK];

    const int t    = threadIdx.x;
    const int wave = t >> 5;
    const int lane = t & 31;
    const int hf   = lane >> 4;      // half-wave 0/1
    const int l16  = lane & 15;
    const int dh   = hf * 8;         // lane-half K offset inside fragments
    const int rowbase = blockIdx.x * ROWS_PER_BLK + wave * 16;

    // Issue the async L2->LDS copy of one 64-row chunk (hi+lo planes).
    // 64 rows * 8 (16B segs) * 2 planes = 1024 segs -> 2 per thread.
    auto issue_copy = [&](int chunk, int buf) {
#pragma unroll
        for (int it = 0; it < 2; ++it) {
            int gid   = it * BLK_THREADS + t;    // 0..1023
            int plane = gid >> 9;                // 0: hi, 1: lo
            int rem   = gid & 511;
            int row   = rem >> 3;                // 0..63
            int seg   = rem & 7;                 // 16B segment of the 64 real elems
            const unsigned short* gsrc =
                (plane ? g_lo : g_hi) +
                (size_t)(chunk * CHUNK_K + row) * ROWSTRIDE + seg * 8;
            unsigned short* lp =
                (plane ? &sh_lo[buf][0] : &sh_hi[buf][0]) + row * ROWSTRIDE + seg * 8;
            unsigned lds_off = (unsigned)(size_t)lp;   // low 32 bits = LDS offset
            asm volatile("global_load_async_to_lds_b128 %0, %1, off"
                         :: "v"(lds_off), "v"(gsrc) : "memory");
        }
        if (t < CHUNK_K) sh_csq[buf][t] = csq[chunk * CHUNK_K + t];
    };

    // A fragments (loop-invariant): z row per lane, hi/lo, both 32-wide K windows
    const float* zrow = z + (size_t)(rowbase + l16) * D_DIM;
    Frag ahi0, alo0, ahi1, alo1;
    build_a_frag(zrow, 0,  dh, ahi0, alo0);
    build_a_frag(zrow, 32, dh, ahi1, alo1);

    float minval[8];
    int   minidx[8];
#pragma unroll
    for (int r = 0; r < 8; ++r) { minval[r] = 3.4e38f; minidx[r] = 0; }

    // Pending-tile pipeline state (argmin consumed one tile behind WMMAs).
    // Dummy priming tile: pcs (3.0e38) < minval-init, so it writes once and is
    // immediately overtaken by any real distance -> argmin unaffected.
    v8f   p0 = {}, p1 = {};
    float pcs   = 3.0e38f;
    int   pkidx = 0;

    auto argmin_update = [&]() {
#pragma unroll
        for (int r = 0; r < 8; ++r) {
            float dist = __builtin_fmaf(-2.0f, p0[r],
                         __builtin_fmaf(-2.0f, p1[r], pcs));
            if (dist < minval[r]) { minval[r] = dist; minidx[r] = pkidx; }
        }
    };

    issue_copy(0, 0);
    for (int c = 0; c < NCHUNK; ++c) {
        const int buf = c & 1;
        if (c + 1 < NCHUNK) {
            issue_copy(c + 1, buf ^ 1);
            // async loads complete in order: <=2 outstanding => chunk c landed,
            // chunk c+1's two transfers stay in flight under the WMMAs.
            asm volatile("s_wait_asynccnt 0x2" ::: "memory");
        } else {
            asm volatile("s_wait_asynccnt 0x0" ::: "memory");
        }
        __syncthreads();    // chunk c visible to all waves

        const unsigned short* ph = &sh_hi[buf][0];
        const unsigned short* pl = &sh_lo[buf][0];
#pragma unroll
        for (int kt = 0; kt < CHUNK_K / 16; ++kt) {
            int kc = kt * 16 + l16;   // B column = codebook row within chunk
            Frag bh0 = load_b_frag(ph, kc, 0,  dh);
            Frag bl0 = load_b_frag(pl, kc, 0,  dh);
            Frag bh1 = load_b_frag(ph, kc, 32, dh);
            Frag bl1 = load_b_frag(pl, kc, 32, dh);
            v8f acc0 = {}, acc1 = {};   // two chains: hi*hi and cross terms
            acc0 = __builtin_amdgcn_wmma_f32_16x16x32_bf16(false, ahi0.bf, false, bh0.bf, (short)0, acc0, false, false);
            acc1 = __builtin_amdgcn_wmma_f32_16x16x32_bf16(false, alo0.bf, false, bh0.bf, (short)0, acc1, false, false);
            acc1 = __builtin_amdgcn_wmma_f32_16x16x32_bf16(false, ahi0.bf, false, bl0.bf, (short)0, acc1, false, false);
            acc0 = __builtin_amdgcn_wmma_f32_16x16x32_bf16(false, ahi1.bf, false, bh1.bf, (short)0, acc0, false, false);
            acc1 = __builtin_amdgcn_wmma_f32_16x16x32_bf16(false, alo1.bf, false, bh1.bf, (short)0, acc1, false, false);
            acc1 = __builtin_amdgcn_wmma_f32_16x16x32_bf16(false, ahi1.bf, false, bl1.bf, (short)0, acc1, false, false);

            // Consume the PREVIOUS tile while this tile's WMMAs are in flight.
            argmin_update();

            p0 = acc0; p1 = acc1;
            pcs   = sh_csq[buf][kt * 16 + l16];
            pkidx = c * CHUNK_K + kt * 16 + l16;
        }
        __syncthreads();    // all waves done reading buf before it is re-filled
    }
    argmin_update();        // drain the final pending tile

    // Reduce argmin across the 16 columns of each half-wave (first-min ties)
#pragma unroll
    for (int m = 1; m <= 8; m <<= 1) {
#pragma unroll
        for (int r = 0; r < 8; ++r) {
            float ov = __shfl_xor(minval[r], m, 32);
            int   oi = __shfl_xor(minidx[r], m, 32);
            if (ov < minval[r] || (ov == minval[r] && oi < minidx[r])) {
                minval[r] = ov; minidx[r] = oi;
            }
        }
    }
    if (l16 == 0) {   // C-tile rows: lanes<16 hold rows r, lanes>=16 rows r+8
#pragma unroll
        for (int r = 0; r < 8; ++r) sh_idx[wave * 16 + hf * 8 + r] = minidx[r];
    }
    __syncthreads();

    // Epilogue: gather codebook[idx], write z_q_st / indices / z_q, MSE partial
    const size_t ND = (size_t)N_TOT * D_DIM;
    float* out_st  = out;
    float* out_idx = out + ND + 1;
    float* out_zq  = out + ND + 1 + N_TOT;

    int row  = rowbase + l16;
    int cidx = sh_idx[wave * 16 + l16];
    if (hf == 0) out_idx[row] = (float)cidx;

    const float* crow = cb + (size_t)cidx * D_DIM + hf * 32;   // 2 lanes per row
    const float* zr2  = z  + (size_t)row  * D_DIM + hf * 32;
    float lsum = 0.f;
#pragma unroll
    for (int j = 0; j < 32; j += 4) {
        float4 c4 = *(const float4*)(crow + j);
        float4 z4 = *(const float4*)(zr2 + j);
        size_t o = (size_t)row * D_DIM + hf * 32 + j;
        *(float4*)(out_st + o) = c4;
        *(float4*)(out_zq + o) = c4;
        float dx = c4.x - z4.x, dy = c4.y - z4.y, dz = c4.z - z4.z, dw = c4.w - z4.w;
        lsum += dx * dx + dy * dy + dz * dz + dw * dw;
    }
#pragma unroll
    for (int m = 1; m < 32; m <<= 1) lsum += __shfl_xor(lsum, m, 32);
    if (lane == 0) sh_wsum[wave] = lsum;
    __syncthreads();
    if (t == 0) {
        float s = 0.f;
        for (int w = 0; w < WAVES_PER_BLK; ++w) s += sh_wsum[w];
        partials[blockIdx.x] = s;
    }
}

// Kernel 2: deterministic fixed-order finalize of vq_loss
__global__ void vq_finalize_kernel(const float* __restrict__ partials,
                                   float* __restrict__ out) {
    int lane = threadIdx.x;                 // 32 threads
    float s = 0.f;
    for (int i = 0; i < NBLK / 32; ++i) s += partials[lane + i * 32];
#pragma unroll
    for (int m = 1; m < 32; m <<= 1) s += __shfl_xor(s, m, 32);
    if (lane == 0) {
        const float ND = 16777216.0f;       // N*D
        out[(size_t)N_TOT * D_DIM] = 1.25f * (s / ND);  // (1 + BETA) * MSE
    }
}

extern "C" void kernel_launch(void* const* d_in, const int* in_sizes, int n_in,
                              void* d_out, int out_size, void* d_ws, size_t ws_size,
                              hipStream_t stream) {
    const float* z  = (const float*)d_in[0];   // [N, 64] fp32
    const float* cb = (const float*)d_in[1];   // [1024, 64] fp32
    float* out      = (float*)d_out;

    // ws layout: csq[1024] f32 | partials[1024] f32 | hi plane | lo plane
    float* csq      = (float*)d_ws;
    float* partials = csq + K_CB;
    unsigned short* g_hi = (unsigned short*)((char*)d_ws + 8192);
    unsigned short* g_lo = g_hi + (size_t)K_CB * ROWSTRIDE;   // ~303 KB total

    vq_prep_kernel<<<(K_CB + 255) / 256, 256, 0, stream>>>(cb, csq, g_hi, g_lo);
    vq_main_kernel<<<NBLK, BLK_THREADS, 0, stream>>>(z, cb, csq, g_hi, g_lo, out, partials);
    vq_finalize_kernel<<<1, 32, 0, stream>>>(partials, out);
}
